// MultiHeadedAttention_19207093748651
// MI455X (gfx1250) — matrix operand order repaired and checked by reading, compiled, not run on hardware
//
#include <hip/hip_runtime.h>
#include <hip/hip_bf16.h>
#include <math.h>

// ---------------------------------------------------------------------------
// MultiHeadedAttention with sparse (offsets {5,3,1,0}) attention pattern.
// ds=8, nb=4, S=512, D_MODEL=256, H=8, D_K=32 -> 16384 tokens.
// Memory-bound on MI455X (23.3 TB/s): single pass per tensor, f16 WMMA with
// f32 accumulation (v_wmma_f32_16x16x32_f16), weights L2-resident.
// ---------------------------------------------------------------------------

typedef __attribute__((ext_vector_type(16))) _Float16 v16h;
typedef __attribute__((ext_vector_type(8)))  float    v8f;

#define TOKENS 16384
#define DMODEL 256
#define SEQ    512
#define NHEAD  8
#define DK     32

// Y[m,n] = sum_k A[m,k] * W[n,k] + bias[n]    (i.e. Y = A @ W^T + b)
// Wave tile: 16 rows x 64 cols (4 WMMA N-tiles). Block: 8 waves -> 128 x 64.
__global__ __launch_bounds__(256) void gemm256_f16wmma(
    const float* __restrict__ A, const float* __restrict__ W,
    const float* __restrict__ bias, float* __restrict__ Y)
{
    const int lane    = threadIdx.x & 31;
    const int wave    = threadIdx.x >> 5;
    const int laneHi  = lane >> 4;     // 0: lanes 0-15, 1: lanes 16-31
    const int l16     = lane & 15;
    const int rowTile = blockIdx.x * 128 + wave * 16;
    const int colBase = blockIdx.y * 64;

    const float* Arow = A + (size_t)(rowTile + l16) * DMODEL;

    v8f acc[4];
    #pragma unroll
    for (int t = 0; t < 4; ++t) acc[t] = (v8f){};

    #pragma unroll
    for (int k0 = 0; k0 < DMODEL; k0 += 32) {
        if (k0 + 32 < DMODEL) {
            // pull next K slab of A toward the caches (global_prefetch_b8)
            __builtin_prefetch(Arow + k0 + 32, 0, 0);
        }

        // ---- A fragment: 16x32 f16, ISA layout ----
        // lane (0-15): row = l16; VGPR0-3 hold K = k0+half*8 .. +7,
        //              VGPR4-7 hold K = k0+16+half*8 .. +7   (half = laneHi)
        v16h afrag;
        {
            const float* p0 = Arow + k0 + laneHi * 8;
            const float* p1 = p0 + 16;
            #pragma unroll
            for (int i = 0; i < 8; ++i) afrag[i]     = (_Float16)p0[i];
            #pragma unroll
            for (int i = 0; i < 8; ++i) afrag[8 + i] = (_Float16)p1[i];
        }

        // ---- B fragments: 32x16 f16 per N-tile; B[k,n] = W[n,k] so each
        // lane reads two runs of 8 contiguous floats of row n of W (free transpose).
        // elements 0-7: K = k0+laneHi*8..+7; elements 8-15: K = k0+16+laneHi*8..+7 ----
        #pragma unroll
        for (int t = 0; t < 4; ++t) {
            const int n = colBase + t * 16 + l16;
            const float* wp = W + (size_t)n * DMODEL + k0 + laneHi * 8;
            v16h bfrag;
            #pragma unroll
            for (int i = 0; i < 16; ++i) bfrag[i] = (_Float16)wp[(i & 7) + ((i >> 3) << 4)];

            acc[t] = __builtin_amdgcn_wmma_f32_16x16x32_f16(
                /*neg_a=*/false, afrag, /*neg_b=*/false, bfrag,
                /*c_mod=*/(short)0, acc[t],
                /*reuse_a=*/false, /*reuse_b=*/false);
        }
    }

    // ---- store: C/D layout: VGPR r -> (M = r + laneHi*8, N = l16) ----
    #pragma unroll
    for (int t = 0; t < 4; ++t) {
        const int col = colBase + t * 16 + l16;
        const float bv = bias[col];
        #pragma unroll
        for (int r = 0; r < 8; ++r) {
            const int row = rowTile + r + laneHi * 8;
            Y[(size_t)row * DMODEL + col] = acc[t][r] + bv;
        }
    }
}

// One wave per token; lane = d_k dim; 8 heads sequential.
// Active key offsets for row s: c in {5,3,1,0} with c <= s.
__global__ __launch_bounds__(256) void sparse_attn(
    const float* __restrict__ Q, const float* __restrict__ K,
    const float* __restrict__ V, float* __restrict__ X)
{
    const int lane  = threadIdx.x & 31;
    const int wave  = threadIdx.x >> 5;
    const int token = blockIdx.x * 8 + wave;
    const int s     = token & (SEQ - 1);
    const int seq0  = token - s;               // first token of this sequence
    const int cons[4] = {5, 3, 1, 0};

    #pragma unroll
    for (int h = 0; h < NHEAD; ++h) {
        const int off = h * DK + lane;
        const float qv = Q[(size_t)token * DMODEL + off];

        float sc[4];
        float m = -1e30f;
        #pragma unroll
        for (int c = 0; c < 4; ++c) {
            const bool act = (cons[c] <= s);          // uniform across wave
            int ks = s - cons[c]; if (ks < 0) ks = 0;
            float p = qv * K[(size_t)(seq0 + ks) * DMODEL + off];
            #pragma unroll
            for (int o = 16; o > 0; o >>= 1) p += __shfl_xor(p, o, 32);
            sc[c] = p * 0.5f;                         // / sqrt(n_att=4)
            if (act && sc[c] > m) m = sc[c];
        }

        float sum = 0.0f, out = 0.0f;
        #pragma unroll
        for (int c = 0; c < 4; ++c) {
            if (cons[c] <= s) {                        // uniform branch
                const float w = __expf(sc[c] - m);
                sum += w;
                const int ks = s - cons[c];
                out += w * V[(size_t)(seq0 + ks) * DMODEL + off];
            }
        }
        X[(size_t)token * DMODEL + off] = out / sum;
    }
}

extern "C" void kernel_launch(void* const* d_in, const int* in_sizes, int n_in,
                              void* d_out, int out_size, void* d_ws, size_t ws_size,
                              hipStream_t stream) {
    const float* query  = (const float*)d_in[0];
    const float* key_in = (const float*)d_in[1];
    const float* value  = (const float*)d_in[2];
    const float* Wq = (const float*)d_in[3];
    const float* bq = (const float*)d_in[4];
    const float* Wk = (const float*)d_in[5];
    const float* bk = (const float*)d_in[6];
    const float* Wv = (const float*)d_in[7];
    const float* bv = (const float*)d_in[8];
    const float* Wo = (const float*)d_in[9];
    const float* bo = (const float*)d_in[10];

    const size_t mat = (size_t)TOKENS * DMODEL;   // 4.19M floats = 16 MB
    float* Qs = (float*)d_ws;
    float* Ks = Qs + mat;
    float* Vs = Ks + mat;
    float* Xs = Vs + mat;

    dim3 gblk(256);
    dim3 ggrid(TOKENS / 128, DMODEL / 64, 1);     // 128 x 4 blocks

    gemm256_f16wmma<<<ggrid, gblk, 0, stream>>>(query,  Wq, bq, Qs);
    gemm256_f16wmma<<<ggrid, gblk, 0, stream>>>(key_in, Wk, bk, Ks);
    gemm256_f16wmma<<<ggrid, gblk, 0, stream>>>(value,  Wv, bv, Vs);

    sparse_attn<<<TOKENS / 8, 256, 0, stream>>>(Qs, Ks, Vs, Xs);

    gemm256_f16wmma<<<ggrid, gblk, 0, stream>>>(Xs, Wo, bo, (float*)d_out);
}
